// BertSelfAttention_38680475468056
// MI455X (gfx1250) — compile-verified, matrix operand order checked by reading, not run on hardware
//
#include <hip/hip_runtime.h>
#include <hip/hip_bf16.h>

// ---------------- types ----------------
typedef __attribute__((ext_vector_type(16))) __bf16 v16bf;
typedef __attribute__((ext_vector_type(8)))  __bf16 v8bf;
typedef __attribute__((ext_vector_type(4)))  __bf16 v4bf;
typedef __attribute__((ext_vector_type(8)))  float  v8f;
typedef __attribute__((ext_vector_type(4)))  unsigned int v4u;
typedef __attribute__((ext_vector_type(8)))  int    v8i_;
typedef __attribute__((ext_vector_type(4)))  int    v4i_;

#define WMMA_BF16(a, b, c) \
  __builtin_amdgcn_wmma_f32_16x16x32_bf16(false, (a), false, (b), (short)0, (c), false, false)

// ---------------- problem constants ----------------
constexpr int HID   = 1024;
constexpr int SEQ   = 2048;
constexpr int BATCH = 2;
constexpr int NHEAD = 16;
constexpr int DHEAD = 64;
constexpr int MTOT  = BATCH * SEQ;      // 4096 rows
constexpr float EPSV = 1e-12f;

// ---------------- CDNA5 feature gates (device pass only) ----------------
#if __has_builtin(__builtin_amdgcn_global_load_async_to_lds_b128)
#define USE_ASYNC 1
#else
#define USE_ASYNC 0
#endif

#if __has_builtin(__builtin_amdgcn_tensor_load_to_lds)
#define USE_TDM 1
#else
#define USE_TDM 0
#endif

__device__ __forceinline__ void wait_async0() {
#if __has_builtin(__builtin_amdgcn_s_wait_asynccnt)
  __builtin_amdgcn_s_wait_asynccnt(0);
#else
  asm volatile("s_wait_asynccnt 0x0" ::: "memory");
#endif
}

__device__ __forceinline__ void wait_tensor0() {
#if __has_builtin(__builtin_amdgcn_s_wait_tensorcnt)
  __builtin_amdgcn_s_wait_tensorcnt(0);
#else
  asm volatile("s_wait_tensorcnt 0x0" ::: "memory");
#endif
}

// 16B global -> LDS copy; async DMA when available (ASYNCcnt), else VGPR bounce.
__device__ __forceinline__ void copy16(const __bf16* g, __bf16* l) {
#if USE_ASYNC
  __builtin_amdgcn_global_load_async_to_lds_b128(
      (__attribute__((address_space(1))) v4i_*)(void*)g,
      (__attribute__((address_space(3))) v4i_*)(void*)l, 0, 0);
#else
  *(v8bf*)l = *(const v8bf*)g;
#endif
}

#if USE_TDM
__device__ __forceinline__ unsigned lds_off_of(const __bf16* p) {
  return (unsigned)(uintptr_t)(__attribute__((address_space(3))) const __bf16*)p;
}
// 2-D TDM tile load: tile rows x cols of bf16, global row stride (elements).
__device__ __forceinline__ void tdm_load_tile_2d(unsigned lds_off, const __bf16* gaddr,
                                                 int rows, int cols, int strideElems) {
  const unsigned long long ga = (unsigned long long)(uintptr_t)gaddr;
  v4u g0 = {};
  g0[0] = 1u;                                          // count = 1 (valid)
  g0[1] = lds_off;                                     // lds_addr
  g0[2] = (unsigned)(ga & 0xffffffffu);                // global_addr[31:0]
  g0[3] = (unsigned)((ga >> 32) & 0x01ffffffu) | (2u << 30);  // addr[56:32] | type=2
  v8i_ g1 = {};
  g1[0] = (1 << 16);                                   // data_size = 1 (2 bytes)
  g1[1] = (cols & 0xffff) << 16;                       // tensor_dim0 [63:48]
  g1[2] = ((cols >> 16) & 0xffff) | ((rows & 0xffff) << 16);  // dim0 hi | tensor_dim1 lo
  g1[3] = ((rows >> 16) & 0xffff) | ((cols & 0xffff) << 16);  // dim1 hi | tile_dim0
  g1[4] = (rows & 0xffff);                             // tile_dim1 (tile_dim2 = 0)
  g1[5] = strideElems;                                 // tensor_dim0_stride [191:160]
  g1[6] = 0; g1[7] = 0;
  v4i_ z4 = {};
#if __clang_major__ >= 23
  v8i_ z8 = {};
  __builtin_amdgcn_tensor_load_to_lds(g0, g1, z4, z4, z8, 0);
#else
  __builtin_amdgcn_tensor_load_to_lds(g0, g1, z4, z4, 0);
#endif
}
#endif  // USE_TDM

// A/B operand builder: two contiguous 16B LDS chunks per lane.
// 16-bit A layout (16x32): lane -> row, half = lane/16; K = {half*8..+7} U {16+half*8..+7}.
__device__ __forceinline__ v16bf load16s(const __bf16* lds, int row, int half,
                                         int stride, int kbase) {
  const v8bf lo = *(const v8bf*)&lds[row * stride + kbase + half * 8];
  const v8bf hi = *(const v8bf*)&lds[row * stride + kbase + 16 + half * 8];
  return __builtin_shufflevector(lo, hi, 0, 1, 2, 3, 4, 5, 6, 7,
                                         8, 9, 10, 11, 12, 13, 14, 15);
}

// ---------------- kernel 1: fp32 -> bf16 ----------------
__global__ void cvt_f32_to_bf16(const float* __restrict__ src,
                                __bf16* __restrict__ dst, int n4) {
  int i = blockIdx.x * blockDim.x + threadIdx.x;
  if (i < n4) {
    const float4 f = ((const float4*)src)[i];
    v4bf o;
    o[0] = (__bf16)f.x; o[1] = (__bf16)f.y; o[2] = (__bf16)f.z; o[3] = (__bf16)f.w;
    ((v4bf*)dst)[i] = o;
  }
}

// ---------------- kernel 2: QKV projection GEMM (double-buffered, async) ----------------
// Y[z] = X @ W[z]^T + bias[z]; grid (N/64, M/64, 3), block 128 (4 waves).
__global__ __launch_bounds__(128)
void qkv_gemm(const __bf16* __restrict__ Xb, const __bf16* __restrict__ Wb,
              const float* __restrict__ bq, const float* __restrict__ bk,
              const float* __restrict__ bv, __bf16* __restrict__ QKVb) {
  __shared__ __bf16 ldsX[2][64 * 32];
  __shared__ __bf16 ldsW[2][64 * 32];

  const int n0 = blockIdx.x * 64;
  const int m0 = blockIdx.y * 64;
  const int z  = blockIdx.z;
  const __bf16* W    = Wb + (size_t)z * HID * HID;
  const float*  bias = (z == 0) ? bq : (z == 1) ? bk : bv;
  __bf16*       Y    = QKVb + (size_t)z * MTOT * HID;

  const int t = threadIdx.x;
  const int lane = t & 31, wv = t >> 5;
  const int rlo = lane & 15, half = lane >> 4;

  v8f acc[4] = {};

  // prologue: stage k0 = 0 into buffer 0
  for (int c = t; c < 256; c += 128) {
    const int row = c >> 2, col8 = (c & 3) * 8;
    copy16(&Xb[(size_t)(m0 + row) * HID + col8], &ldsX[0][row * 32 + col8]);
    copy16(&W[(size_t)(n0 + row) * HID + col8], &ldsW[0][row * 32 + col8]);
  }
  wait_async0();
  __syncthreads();

  int cur = 0;
  for (int k0 = 0; k0 < HID; k0 += 32) {
    const int nxt = cur ^ 1;
    if (k0 + 32 < HID) {  // issue next tiles while computing current
      for (int c = t; c < 256; c += 128) {
        const int row = c >> 2, col8 = (c & 3) * 8;
        copy16(&Xb[(size_t)(m0 + row) * HID + k0 + 32 + col8], &ldsX[nxt][row * 32 + col8]);
        copy16(&W[(size_t)(n0 + row) * HID + k0 + 32 + col8], &ldsW[nxt][row * 32 + col8]);
      }
    }

    const v16bf a = load16s(ldsX[cur], wv * 16 + rlo, half, 32, 0);
#pragma unroll
    for (int ns = 0; ns < 4; ++ns) {
      const v16bf bm = load16s(ldsW[cur], ns * 16 + rlo, half, 32, 0);
      acc[ns] = WMMA_BF16(a, bm, acc[ns]);
    }

    wait_async0();
    __syncthreads();
    cur = nxt;
  }

#pragma unroll
  for (int ns = 0; ns < 4; ++ns) {
    const int n = n0 + ns * 16 + rlo;
    const float bz = bias[n];
#pragma unroll
    for (int i = 0; i < 8; ++i) {
      const int m = m0 + wv * 16 + half * 8 + i;
      Y[(size_t)m * HID + n] = (__bf16)(acc[ns][i] + bz);
    }
  }
}

// ---------------- kernel 3: fused relu-softmax attention ----------------
// grid (SEQ/64, NHEAD, BATCH), block 128 (4 waves, 16 q-rows each).
// Q via TDM; K double-buffered async->LDS; V pipelined through regs into LDS^T.
__global__ __launch_bounds__(128)
void attn_fused(const __bf16* __restrict__ Qb, const __bf16* __restrict__ Kb,
                const __bf16* __restrict__ Vb, const float* __restrict__ mask,
                float* __restrict__ out) {
  __shared__ __bf16 ldsQ[64 * 64];
  __shared__ __bf16 ldsK[2][64 * 64];
  __shared__ __bf16 ldsVT[2][64 * 64];   // V transposed: [d][key]
  __shared__ __bf16 ldsP[64 * 64];       // relu scores, bf16
  __shared__ float rowsumL[64];
  __shared__ float colsumL[64];

  const int q0 = blockIdx.x * 64;
  const int h  = blockIdx.y;
  const int b  = blockIdx.z;
  const int t = threadIdx.x, lane = t & 31, wv = t >> 5;
  const int rlo = lane & 15, half = lane >> 4;
  const size_t rowbase = (size_t)b * SEQ;
  const int cbase = h * DHEAD;

  // ---- Q tile [64 x 64]: TDM strided 2-D DMA (wave 0 issues once) ----
#if USE_TDM
  if (wv == 0) {
    tdm_load_tile_2d(lds_off_of(&ldsQ[0]), &Qb[(rowbase + q0) * HID + cbase],
                     64, 64, HID);
  }
#else
  for (int c = t; c < 512; c += 128) {
    const int row = c >> 3, col8 = (c & 7) * 8;
    *(v8bf*)&ldsQ[row * 64 + col8] =
        *(const v8bf*)&Qb[(rowbase + q0 + row) * HID + cbase + col8];
  }
#endif

  // ---- prologue: stage key-tile 0 into buffer 0 ----
  v8bf vreg[4];
  for (int c = t, j = 0; c < 512; c += 128, ++j) {
    const int row = c >> 3, col8 = (c & 7) * 8;
    const size_t g = (rowbase + row) * HID + cbase + col8;
    copy16(&Kb[g], &ldsK[0][row * 64 + col8]);
    vreg[j] = *(const v8bf*)&Vb[g];
  }
  for (int c = t, j = 0; c < 512; c += 128, ++j) {
    const int row = c >> 3, col8 = (c & 7) * 8;
#pragma unroll
    for (int jj = 0; jj < 8; ++jj) ldsVT[0][(col8 + jj) * 64 + row] = vreg[j][jj];
  }
  wait_async0();
#if USE_TDM
  if (wv == 0) wait_tensor0();
#endif
  __syncthreads();

  v8f ctx[4] = {};
  float rowsum = 0.f;
  float colsum = 0.f;   // threads t<64: column t of V
  const float scale = 0.125f;  // 1/sqrt(64)

  int cur = 0;
  for (int k0 = 0; k0 < SEQ; k0 += 64) {
    const int nxt = cur ^ 1;
    const bool more = (k0 + 64) < SEQ;

    if (more) {  // issue next K tile (async DMA) + next V tile into registers
      for (int c = t, j = 0; c < 512; c += 128, ++j) {
        const int row = c >> 3, col8 = (c & 7) * 8;
        const size_t g = (rowbase + k0 + 64 + row) * HID + cbase + col8;
        copy16(&Kb[g], &ldsK[nxt][row * 64 + col8]);
        vreg[j] = *(const v8bf*)&Vb[g];
      }
    }

    if (t < 64) {  // colsum(V) for the eps/S term
      float cs = 0.f;
      for (int r = 0; r < 64; ++r) cs += (float)ldsVT[cur][t * 64 + r];
      colsum += cs;
    }

    // S = Q K^T
    v8f sc[4] = {};
    {
      const int rq = wv * 16 + rlo;
#pragma unroll
      for (int kk = 0; kk < 2; ++kk) {
        const v16bf a = load16s(ldsQ, rq, half, 64, kk * 32);
#pragma unroll
        for (int ns = 0; ns < 4; ++ns) {
          const v16bf bm = load16s(ldsK[cur], ns * 16 + rlo, half, 64, kk * 32);
          sc[ns] = WMMA_BF16(a, bm, sc[ns]);
        }
      }
    }

    // scale + additive mask + relu -> P
#pragma unroll
    for (int ns = 0; ns < 4; ++ns) {
      const int keyLocal = ns * 16 + rlo;
      const float mk = mask[rowbase + k0 + keyLocal];
#pragma unroll
      for (int i = 0; i < 8; ++i) {
        float s = sc[ns][i] * scale + mk;
        s = s > 0.f ? s : 0.f;
        ldsP[(wv * 16 + half * 8 + i) * 64 + keyLocal] = (__bf16)s;
      }
    }
    __syncthreads();

    // ctx += P @ V ; row sums ride along on the A-layout reads of P
    {
      const int rp = wv * 16 + rlo;
      float part = 0.f;
#pragma unroll
      for (int kk = 0; kk < 2; ++kk) {
        const v16bf a = load16s(ldsP, rp, half, 64, kk * 32);
#pragma unroll
        for (int e = 0; e < 16; ++e) part += (float)a[e];
#pragma unroll
        for (int dsb = 0; dsb < 4; ++dsb) {
          const v16bf bm = load16s(ldsVT[cur], dsb * 16 + rlo, half, 64, kk * 32);
          ctx[dsb] = WMMA_BF16(a, bm, ctx[dsb]);
        }
      }
      part += __shfl_xor(part, 16, 32);
      rowsum += part;
    }

    if (more) {  // retire pipelined V regs into the next transposed buffer
      for (int c = t, j = 0; c < 512; c += 128, ++j) {
        const int row = c >> 3, col8 = (c & 7) * 8;
#pragma unroll
        for (int jj = 0; jj < 8; ++jj)
          ldsVT[nxt][(col8 + jj) * 64 + row] = vreg[j][jj];
      }
      wait_async0();
    }
    __syncthreads();
    cur = nxt;
  }

  if (half == 0) rowsumL[wv * 16 + rlo] = rowsum;
  if (t < 64) colsumL[t] = colsum;
  __syncthreads();

  const float epsS = EPSV / (float)SEQ;
#pragma unroll
  for (int dsb = 0; dsb < 4; ++dsb) {
    const int d = dsb * 16 + rlo;
    const float csv = colsumL[d] * epsS;
#pragma unroll
    for (int i = 0; i < 8; ++i) {
      const int rloc = wv * 16 + half * 8 + i;
      const float denom = rowsumL[rloc] + EPSV;
      out[(rowbase + q0 + rloc) * HID + cbase + d] = (ctx[dsb][i] + csv) / denom;
    }
  }
}

// ---------------- launcher ----------------
extern "C" void kernel_launch(void* const* d_in, const int* in_sizes, int n_in,
                              void* d_out, int out_size, void* d_ws, size_t ws_size,
                              hipStream_t stream) {
  const float* hidden = (const float*)d_in[0];
  const float* mask   = (const float*)d_in[1];
  const float* Wq = (const float*)d_in[2];
  const float* bq = (const float*)d_in[3];
  const float* Wk = (const float*)d_in[4];
  const float* bk = (const float*)d_in[5];
  const float* Wv = (const float*)d_in[6];
  const float* bv = (const float*)d_in[7];
  float* out = (float*)d_out;

  __bf16* Xb   = (__bf16*)d_ws;                      // 4096*1024
  __bf16* Wb   = Xb + (size_t)MTOT * HID;            // 3*1024*1024
  __bf16* QKVb = Wb + (size_t)3 * HID * HID;         // 3*4096*1024

  {
    const int n4 = MTOT * HID / 4;
    cvt_f32_to_bf16<<<(n4 + 255) / 256, 256, 0, stream>>>(hidden, Xb, n4);
  }
  {
    const int n4 = HID * HID / 4;
    const int g = (n4 + 255) / 256;
    cvt_f32_to_bf16<<<g, 256, 0, stream>>>(Wq, Wb + (size_t)0 * HID * HID, n4);
    cvt_f32_to_bf16<<<g, 256, 0, stream>>>(Wk, Wb + (size_t)1 * HID * HID, n4);
    cvt_f32_to_bf16<<<g, 256, 0, stream>>>(Wv, Wb + (size_t)2 * HID * HID, n4);
  }

  qkv_gemm<<<dim3(HID / 64, MTOT / 64, 3), 128, 0, stream>>>(Xb, Wb, bq, bk, bv, QKVb);

  attn_fused<<<dim3(SEQ / 64, NHEAD, BATCH), 128, 0, stream>>>(
      QKVb + (size_t)0 * MTOT * HID,
      QKVb + (size_t)1 * MTOT * HID,
      QKVb + (size_t)2 * MTOT * HID,
      mask, out);
}
